// H2GT_11235634446348
// MI455X (gfx1250) — matrix operand order repaired and checked by reading, not compiled, not run on hardware
//
#include <hip/hip_runtime.h>
#include <hip/hip_bf16.h>

typedef __attribute__((ext_vector_type(16))) _Float16 v16h;
typedef __attribute__((ext_vector_type(8)))  _Float16 v8h;
typedef __attribute__((ext_vector_type(8)))  float    v8f;

#define NN     4096
#define DD     256
#define MM     4100
#define MP     4128     // M padded to a multiple of 32 for attention tiling
#define MW     132      // mask words per node row (MP/32 = 129, padded to 132)
#define ALPHA  0.5f
#define LN_EPS 1e-5f
#define NEG_BIG -1e30f

// ---------------------------------------------------------------------------
// WMMA helpers (CDNA5 v_wmma_f32_16x16x32_f16, wave32). Layouts per ISA 7.12.2
// A (16x32, M across lanes): lane-group grp holds contiguous K run [16g..16g+15] (this code's order)
// B (32x16, N across lanes): same contiguous K run [16g..16g+15] as A (code's order, not ISA's)
// C/D: element r -> row r + 8*grp, col lane&15
// ---------------------------------------------------------------------------
static __device__ __forceinline__ v8f wmma16(v16h a, v16h b, v8f c) {
  return __builtin_amdgcn_wmma_f32_16x16x32_f16(false, a, false, b, (short)0, c,
                                                false, false);
}
// p = row base + k0 + grp*16 ; 16 contiguous halves  (2x b128)
static __device__ __forceinline__ v16h loadA16(const _Float16* p) {
  v8h lo = *(const v8h*)(p);
  v8h hi = *(const v8h*)(p + 8);
  return __builtin_shufflevector(lo, hi, 0, 1, 2, 3, 4, 5, 6, 7,
                                 8, 9, 10, 11, 12, 13, 14, 15);
}
// p = row base + k0 + grp*16 ; 16 contiguous halves  (2x b128)
static __device__ __forceinline__ v16h loadB16(const _Float16* p) {
  v8h lo = *(const v8h*)(p);
  v8h hi = *(const v8h*)(p + 8);
  return __builtin_shufflevector(lo, hi, 0, 1, 2, 3, 4, 5, 6, 7,
                                 8, 9, 10, 11, 12, 13, 14, 15);
}

// ---------------------------------------------------------------------------
// deg[m] = sum_n H[n,m]
// ---------------------------------------------------------------------------
__global__ void deg_kernel(const float* __restrict__ H, float* __restrict__ deg) {
  int m = blockIdx.x * blockDim.x + threadIdx.x;
  if (m >= MM) return;
  float s = 0.f;
  for (int n = 0; n < NN; ++n) s += H[(size_t)n * MM + m];
  deg[m] = s;
}

// ---------------------------------------------------------------------------
// HT[m][n] = f16(H[n][m]), zero-padded to MP rows. LDS-tiled 32x32 transpose.
// grid = dim3(MP/32, NN/32), block = 256 (32x8)
// ---------------------------------------------------------------------------
__global__ void hT_kernel(const float* __restrict__ H, _Float16* __restrict__ HT) {
  __shared__ _Float16 t[32][33];
  const int m0 = blockIdx.x * 32, n0 = blockIdx.y * 32;
  const int tx = threadIdx.x & 31, ty = threadIdx.x >> 5;
#pragma unroll
  for (int j = 0; j < 32; j += 8) {
    int m = m0 + tx;
    int msafe = (m < MM) ? m : (MM - 1);
    float v = H[(size_t)(n0 + ty + j) * MM + msafe];
    t[ty + j][tx] = (_Float16)((m < MM) ? v : 0.f);
  }
  __syncthreads();
#pragma unroll
  for (int j = 0; j < 32; j += 8)
    HT[(size_t)(m0 + ty + j) * NN + n0 + tx] = t[tx][ty + j];
}

// ---------------------------------------------------------------------------
// Bitmask of H: Mb[n][w] bit j = (H[n][w*32+j] != 0), zero beyond MM.
// grid = NN, block = 160
// ---------------------------------------------------------------------------
__global__ void mask_kernel(const float* __restrict__ H, unsigned* __restrict__ Mb) {
  int n = blockIdx.x, w = threadIdx.x;
  if (w >= MW) return;
  unsigned bits = 0;
  for (int j = 0; j < 32; ++j) {
    int m = w * 32 + j;
    if (m < MM && H[(size_t)n * MM + m] != 0.f) bits |= (1u << j);
  }
  Mb[(size_t)n * MW + w] = bits;
}

// ---------------------------------------------------------------------------
// f16 tiled transpose: dst[c][r] = src[r][c]. grid = dim3(C/32, R/32), block 256
// ---------------------------------------------------------------------------
__global__ void t16_kernel(const _Float16* __restrict__ src, _Float16* __restrict__ dst,
                           int R, int C) {
  __shared__ _Float16 t[32][33];
  const int c0 = blockIdx.x * 32, r0 = blockIdx.y * 32;
  const int tx = threadIdx.x & 31, ty = threadIdx.x >> 5;
#pragma unroll
  for (int j = 0; j < 32; j += 8)
    t[ty + j][tx] = src[(size_t)(r0 + ty + j) * C + c0 + tx];
  __syncthreads();
#pragma unroll
  for (int j = 0; j < 32; j += 8)
    dst[(size_t)(c0 + ty + j) * R + r0 + tx] = t[tx][ty + j];
}

// ---------------------------------------------------------------------------
// Per-row LayerNorm + f16 conversion. one block (256 thr) per row
// ---------------------------------------------------------------------------
__global__ void lncvt_kernel(const float* __restrict__ X, _Float16* __restrict__ X16,
                             _Float16* __restrict__ Xn16, const float* __restrict__ g,
                             const float* __restrict__ b) {
  __shared__ float sm[DD];
  int row = blockIdx.x, t = threadIdx.x;
  float x = X[(size_t)row * DD + t];
  sm[t] = x; __syncthreads();
  for (int s = 128; s > 0; s >>= 1) { if (t < s) sm[t] += sm[t + s]; __syncthreads(); }
  float mean = sm[0] * (1.f / DD); __syncthreads();
  float d = x - mean;
  sm[t] = d * d; __syncthreads();
  for (int s = 128; s > 0; s >>= 1) { if (t < s) sm[t] += sm[t + s]; __syncthreads(); }
  float var = sm[0] * (1.f / DD);
  X16[(size_t)row * DD + t] = (_Float16)x;
  Xn16[(size_t)row * DD + t] = (_Float16)(d * rsqrtf(var + LN_EPS) * g[t] + b[t]);
}

__global__ void cvt_kernel(const float* __restrict__ s, _Float16* __restrict__ d, int n) {
  int i = blockIdx.x * blockDim.x + threadIdx.x;
  if (i < n) d[i] = (_Float16)s[i];
}

// ---------------------------------------------------------------------------
// En = LN( (H^T X) / deg ).  Block: 16 hyperedge rows x full 256 cols.
// A = HT rows (vector loads), B = X16T (D x N, vector loads). grid MP/16.
// ---------------------------------------------------------------------------
__global__ void e_kernel(const _Float16* __restrict__ HT,
                         const _Float16* __restrict__ X16T,
                         const float* __restrict__ deg, const float* __restrict__ g,
                         const float* __restrict__ b, _Float16* __restrict__ En16) {
  __shared__ float tile[16][DD + 1];
  __shared__ float mu[16], rs[16];
  const int tid = threadIdx.x, wid = tid >> 5, lane = tid & 31;
  const int grp = lane >> 4, idx = lane & 15;
  const int m0 = blockIdx.x * 16;
  const int n0 = wid * 32;
  const _Float16* arow = HT + (size_t)(m0 + idx) * NN + grp * 16;
  const _Float16* brow0 = X16T + (size_t)(n0 + idx) * NN + grp * 16;
  const _Float16* brow1 = X16T + (size_t)(n0 + 16 + idx) * NN + grp * 16;

  v8f c0 = {}, c1 = {};
  for (int k0 = 0; k0 < NN; k0 += 32) {
    v16h a = loadA16(arow + k0);
    v16h b0 = loadB16(brow0 + k0);
    v16h b1 = loadB16(brow1 + k0);
    c0 = wmma16(a, b0, c0);
    c1 = wmma16(a, b1, c1);
  }
#pragma unroll
  for (int r = 0; r < 8; ++r) {
    int mr = m0 + r + 8 * grp;
    float dv = (mr < MM) ? deg[mr] : 1.f;
    tile[r + 8 * grp][n0 + idx] = c0[r] / dv;
    tile[r + 8 * grp][n0 + 16 + idx] = c1[r] / dv;
  }
  __syncthreads();
  if (tid < 16) {
    float s = 0.f, s2 = 0.f;
    for (int j = 0; j < DD; ++j) { float v = tile[tid][j]; s += v; s2 += v * v; }
    float mean = s * (1.f / DD);
    float var = s2 * (1.f / DD) - mean * mean;
    mu[tid] = mean;
    rs[tid] = rsqrtf(var + LN_EPS);
  }
  __syncthreads();
  for (int r = 0; r < 16; ++r)
    En16[(size_t)(m0 + r) * DD + tid] =
        (_Float16)((tile[r][tid] - mu[r]) * rs[r] * g[tid] + b[tid]);
}

// ---------------------------------------------------------------------------
// C[rows,256] = A16[rows,256] @ W16^T + bias  (W16 row-major [out,in] as in torch;
// B[k][n] = W16[n][k] -> lane reads along a W row: contiguous vector loads)
//   mode 0: store f16   mode 1: relu + residual mix -> f32
//   mode 2: tanh -> f16 mode 3: sigmoid -> f16
// grid = rows/16, block = 256
// ---------------------------------------------------------------------------
__global__ void gemm256_kernel(const _Float16* __restrict__ A,
                               const _Float16* __restrict__ W16,
                               const float* __restrict__ bias, int mode,
                               _Float16* __restrict__ out16, float* __restrict__ out32,
                               const float* __restrict__ prev) {
  const int tid = threadIdx.x, wid = tid >> 5, lane = tid & 31;
  const int grp = lane >> 4, idx = lane & 15;
  const int row0 = blockIdx.x * 16;
  const int n0 = wid * 32;
  const _Float16* arow = A + (size_t)(row0 + idx) * DD + grp * 16;
  const _Float16* brow0 = W16 + (size_t)(n0 + idx) * DD + grp * 16;
  const _Float16* brow1 = W16 + (size_t)(n0 + 16 + idx) * DD + grp * 16;

  v8f c0 = {}, c1 = {};
#pragma unroll
  for (int k0 = 0; k0 < DD; k0 += 32) {
    v16h a = loadA16(arow + k0);
    v16h b0 = loadB16(brow0 + k0);
    v16h b1 = loadB16(brow1 + k0);
    c0 = wmma16(a, b0, c0);
    c1 = wmma16(a, b1, c1);
  }
#pragma unroll
  for (int r = 0; r < 8; ++r) {
    int row = row0 + r + 8 * grp;
#pragma unroll
    for (int t = 0; t < 2; ++t) {
      int col = n0 + 16 * t + idx;
      float v = (t ? c1[r] : c0[r]) + bias[col];
      size_t o = (size_t)row * DD + col;
      if (mode == 0) out16[o] = (_Float16)v;
      else if (mode == 1) { v = fmaxf(v, 0.f); out32[o] = ALPHA * v + (1.f - ALPHA) * prev[o]; }
      else if (mode == 2) out16[o] = (_Float16)tanhf(v);
      else out16[o] = (_Float16)(1.f / (1.f + __expf(-v)));
    }
  }
}

// ---------------------------------------------------------------------------
// k rows 4096..4098 use Wkt, row 4099 uses Wks; rows 4100..4127 zero-padded.
// grid = MP-NN, block = 256
// ---------------------------------------------------------------------------
__global__ void ktail_kernel(const _Float16* __restrict__ En16,
                             const float* __restrict__ Wkt, const float* __restrict__ bkt,
                             const float* __restrict__ Wks, const float* __restrict__ bks,
                             _Float16* __restrict__ k16) {
  int m = NN + blockIdx.x, o = threadIdx.x;
  if (m >= MM) { k16[(size_t)m * DD + o] = (_Float16)0.f; return; }
  const float* W = (m == MM - 1) ? Wks : Wkt;
  const float* bb = (m == MM - 1) ? bks : bkt;
  float acc = bb[o];
  for (int k = 0; k < DD; ++k) acc += (float)En16[(size_t)m * DD + k] * W[o * DD + k];
  k16[(size_t)m * DD + o] = (_Float16)acc;
}

// ---------------------------------------------------------------------------
// Masked flash attention. One wave per (head, 16-node tile); 32-edge chunks:
// 2 WMMA scores (B = k rows, contiguous), bitmask, online softmax, P through
// per-wave LDS (same-wave DS in-order), 2 WMMA PV (B = k16T rows, contiguous).
// grid = 256 (2048 waves), block = 256
// ---------------------------------------------------------------------------
__global__ void attn_kernel(const _Float16* __restrict__ q16,
                            const _Float16* __restrict__ k16,
                            const _Float16* __restrict__ k16T,
                            const unsigned* __restrict__ Mb,
                            _Float16* __restrict__ feat16) {
  __shared__ _Float16 psh[8][16][32];
  const int tid = threadIdx.x, wid = tid >> 5, lane = tid & 31;
  const int grp = lane >> 4, idx = lane & 15;
  const int gw = blockIdx.x * 8 + wid;
  const int head = gw >> 8;
  const int n0 = (gw & 255) * 16;
  const int hoff = head * 32;
  const float sc = 0.17677669529663687f;  // 1/sqrt(32)

  v16h qa = loadA16(q16 + (size_t)(n0 + idx) * DD + hoff + grp * 16);
  const _Float16* kvrow0 = k16T + (size_t)(hoff + idx) * MP + grp * 16;
  const _Float16* kvrow1 = k16T + (size_t)(hoff + 16 + idx) * MP + grp * 16;
  const _Float16* pshrow = &psh[wid][idx][grp * 16];

  v8f acc0 = {}, acc1 = {};
  float mrow[8], lrow[8];
#pragma unroll
  for (int r = 0; r < 8; ++r) { mrow[r] = NEG_BIG; lrow[r] = 0.f; }

  for (int m0 = 0; m0 < MP; m0 += 32) {
    v16h b0 = loadB16(k16 + (size_t)(m0 + idx) * DD + hoff + grp * 16);
    v16h b1 = loadB16(k16 + (size_t)(m0 + 16 + idx) * DD + hoff + grp * 16);
    v8f z = {};
    v8f s0 = wmma16(qa, b0, z);
    v8f s1 = wmma16(qa, b1, z);

#pragma unroll
    for (int r = 0; r < 8; ++r) {
      int nrow = n0 + r + 8 * grp;
      unsigned wbits = Mb[(size_t)nrow * MW + (m0 >> 5)];
      float v0 = ((wbits >> idx) & 1u) ? s0[r] * sc : NEG_BIG;
      float v1 = ((wbits >> (idx + 16)) & 1u) ? s1[r] * sc : NEG_BIG;
      float mx = fmaxf(v0, v1);
      mx = fmaxf(mx, __shfl_xor(mx, 1, 32));
      mx = fmaxf(mx, __shfl_xor(mx, 2, 32));
      mx = fmaxf(mx, __shfl_xor(mx, 4, 32));
      mx = fmaxf(mx, __shfl_xor(mx, 8, 32));   // row max within 16-lane group
      float mnew = fmaxf(mrow[r], mx);
      float scale = __expf(mrow[r] - mnew);
      float e0 = __expf(v0 - mnew), e1 = __expf(v1 - mnew);
      float ts = e0 + e1;
      ts += __shfl_xor(ts, 1, 32);
      ts += __shfl_xor(ts, 2, 32);
      ts += __shfl_xor(ts, 4, 32);
      ts += __shfl_xor(ts, 8, 32);
      lrow[r] = lrow[r] * scale + ts;
      mrow[r] = mnew;
      acc0[r] *= scale; acc1[r] *= scale;
      psh[wid][r + 8 * grp][idx] = (_Float16)e0;
      psh[wid][r + 8 * grp][idx + 16] = (_Float16)e1;
    }
    v16h pa = loadA16(pshrow);               // C-layout -> A-layout via LDS
    v16h kb0 = loadB16(kvrow0 + m0);
    v16h kb1 = loadB16(kvrow1 + m0);
    acc0 = wmma16(pa, kb0, acc0);
    acc1 = wmma16(pa, kb1, acc1);
  }
#pragma unroll
  for (int r = 0; r < 8; ++r) {
    int nrow = n0 + r + 8 * grp;
    float inv = 1.f / lrow[r];
    feat16[(size_t)nrow * DD + hoff + idx] = (_Float16)(acc0[r] * inv);
    feat16[(size_t)nrow * DD + hoff + 16 + idx] = (_Float16)(acc1[r] * inv);
  }
}

// ---------------------------------------------------------------------------
// Araw[n] = dot(a*b, cw) + cb ; one wave per row
// ---------------------------------------------------------------------------
__global__ void gated_score_kernel(const _Float16* __restrict__ a16,
                                   const _Float16* __restrict__ b16,
                                   const float* __restrict__ cw,
                                   const float* __restrict__ cb,
                                   float* __restrict__ Araw) {
  int wid = threadIdx.x >> 5, lane = threadIdx.x & 31;
  int row = blockIdx.x * 8 + wid;
  float s = 0.f;
  for (int d = lane; d < DD; d += 32)
    s += (float)a16[(size_t)row * DD + d] * (float)b16[(size_t)row * DD + d] * cw[d];
  s += __shfl_xor(s, 1, 32); s += __shfl_xor(s, 2, 32); s += __shfl_xor(s, 4, 32);
  s += __shfl_xor(s, 8, 32); s += __shfl_xor(s, 16, 32);
  if (lane == 0) Araw[row] = s + cb[0];
}

// ---------------------------------------------------------------------------
// softmax over N, pooled = w @ X, out = pooled @ out_w^T + out_b. 1 block.
// ---------------------------------------------------------------------------
__global__ void pool_out_kernel(const float* __restrict__ Araw,
                                const float* __restrict__ X,
                                const float* __restrict__ out_w,
                                const float* __restrict__ out_b,
                                float* __restrict__ out) {
  __shared__ float red[DD];
  __shared__ float pooled[DD];
  __shared__ float wsh[NN];
  int t = threadIdx.x;
  float lm = NEG_BIG;
  for (int n = t; n < NN; n += DD) lm = fmaxf(lm, Araw[n]);
  red[t] = lm; __syncthreads();
  for (int s = 128; s > 0; s >>= 1) { if (t < s) red[t] = fmaxf(red[t], red[t + s]); __syncthreads(); }
  float gmax = red[0]; __syncthreads();
  float ls = 0.f;
  for (int n = t; n < NN; n += DD) ls += __expf(Araw[n] - gmax);
  red[t] = ls; __syncthreads();
  for (int s = 128; s > 0; s >>= 1) { if (t < s) red[t] += red[t + s]; __syncthreads(); }
  float inv = 1.f / red[0]; __syncthreads();
  for (int n = t; n < NN; n += DD) wsh[n] = __expf(Araw[n] - gmax) * inv;
  __syncthreads();
  float pd = 0.f;
  for (int n = 0; n < NN; ++n) pd += wsh[n] * X[(size_t)n * DD + t];
  pooled[t] = pd; __syncthreads();
  if (t < 4) {
    float s = out_b[t];
    for (int d = 0; d < DD; ++d) s += pooled[d] * out_w[t * DD + d];
    out[t] = s;
  }
}

// ---------------------------------------------------------------------------
extern "C" void kernel_launch(void* const* d_in, const int* in_sizes, int n_in,
                              void* d_out, int out_size, void* d_ws, size_t ws_size,
                              hipStream_t stream) {
  (void)in_sizes; (void)n_in; (void)out_size; (void)ws_size;
  const float* X    = (const float*)d_in[0];
  const float* H    = (const float*)d_in[1];
  const float* Wq_w = (const float*)d_in[2];  const float* Wq_b = (const float*)d_in[3];
  const float* Wkn_w= (const float*)d_in[4];  const float* Wkn_b= (const float*)d_in[5];
  const float* Wkt_w= (const float*)d_in[6];  const float* Wkt_b= (const float*)d_in[7];
  const float* Wks_w= (const float*)d_in[8];  const float* Wks_b= (const float*)d_in[9];
  const float* fc_w = (const float*)d_in[10]; const float* fc_b = (const float*)d_in[11];
  const float* ln_g = (const float*)d_in[12]; const float* ln_b = (const float*)d_in[13];
  const float* aw   = (const float*)d_in[14]; const float* ab   = (const float*)d_in[15];
  const float* bw   = (const float*)d_in[16]; const float* bb   = (const float*)d_in[17];
  const float* cw   = (const float*)d_in[18]; const float* cb   = (const float*)d_in[19];
  const float* outw = (const float*)d_in[20]; const float* outb = (const float*)d_in[21];

  size_t off = 0;
  char* base = (char*)d_ws;
  auto alloc = [&](size_t bytes) -> void* {
    void* p = base + off; off += (bytes + 255) & ~(size_t)255; return p;
  };
  float*    bufX0  = (float*)alloc((size_t)NN * DD * 4);
  float*    bufX1  = (float*)alloc((size_t)NN * DD * 4);
  _Float16* HT     = (_Float16*)alloc((size_t)MP * NN * 2);
  unsigned* Mb     = (unsigned*)alloc((size_t)NN * MW * 4);
  _Float16* X16    = (_Float16*)alloc((size_t)NN * DD * 2);
  _Float16* X16T   = (_Float16*)alloc((size_t)NN * DD * 2);
  _Float16* Xn16   = (_Float16*)alloc((size_t)NN * DD * 2);
  _Float16* En16   = (_Float16*)alloc((size_t)MP * DD * 2);
  _Float16* q16    = (_Float16*)alloc((size_t)NN * DD * 2);
  _Float16* k16    = (_Float16*)alloc((size_t)MP * DD * 2);
  _Float16* k16T   = (_Float16*)alloc((size_t)MP * DD * 2);
  _Float16* feat16 = (_Float16*)alloc((size_t)NN * DD * 2);
  _Float16* W16a   = (_Float16*)alloc((size_t)DD * DD * 2);
  _Float16* W16b   = (_Float16*)alloc((size_t)DD * DD * 2);
  _Float16* W16c   = (_Float16*)alloc((size_t)DD * DD * 2);
  float*    deg    = (float*)alloc((size_t)MP * 4);
  float*    Araw   = (float*)alloc((size_t)NN * 4);

  // H-derived precomputes (H constant across layers)
  deg_kernel<<<(MM + 255) / 256, 256, 0, stream>>>(H, deg);
  hT_kernel<<<dim3(MP / 32, NN / 32), 256, 0, stream>>>(H, HT);
  mask_kernel<<<NN, 160, 0, stream>>>(H, Mb);

  for (int L = 0; L < 2; ++L) {
    const float* Xcur = L ? bufX0 : X;
    float* Xnext = L ? bufX1 : bufX0;
    size_t wo = (size_t)L * DD * DD, bo = (size_t)L * DD;
    cvt_kernel<<<DD * DD / 256, 256, 0, stream>>>(Wq_w + wo, W16a, DD * DD);
    cvt_kernel<<<DD * DD / 256, 256, 0, stream>>>(Wkn_w + wo, W16b, DD * DD);
    cvt_kernel<<<DD * DD / 256, 256, 0, stream>>>(fc_w + wo, W16c, DD * DD);
    lncvt_kernel<<<NN, DD, 0, stream>>>(Xcur, X16, Xn16, ln_g + bo, ln_b + bo);
    t16_kernel<<<dim3(DD / 32, NN / 32), 256, 0, stream>>>(X16, X16T, NN, DD);
    e_kernel<<<MP / 16, 256, 0, stream>>>(HT, X16T, deg, ln_g + bo, ln_b + bo, En16);
    gemm256_kernel<<<NN / 16, 256, 0, stream>>>(Xn16, W16a, Wq_b + bo, 0, q16, nullptr, nullptr);
    gemm256_kernel<<<NN / 16, 256, 0, stream>>>(En16, W16b, Wkn_b + bo, 0, k16, nullptr, nullptr);
    ktail_kernel<<<MP - NN, DD, 0, stream>>>(En16, Wkt_w + wo, Wkt_b + bo,
                                             Wks_w + wo, Wks_b + bo, k16);
    t16_kernel<<<dim3(DD / 32, MP / 32), 256, 0, stream>>>(k16, k16T, MP, DD);
    attn_kernel<<<256, 256, 0, stream>>>(q16, k16, k16T, Mb, feat16);
    gemm256_kernel<<<NN / 16, 256, 0, stream>>>(feat16, W16c, fc_b + bo, 1, nullptr, Xnext, Xcur);
  }

  // gated attention pooling
  cvt_kernel<<<(NN * DD) / 256, 256, 0, stream>>>(bufX1, X16, NN * DD);
  cvt_kernel<<<DD * DD / 256, 256, 0, stream>>>(aw, W16a, DD * DD);
  cvt_kernel<<<DD * DD / 256, 256, 0, stream>>>(bw, W16b, DD * DD);
  gemm256_kernel<<<NN / 16, 256, 0, stream>>>(X16, W16a, ab, 2, q16, nullptr, nullptr);
  gemm256_kernel<<<NN / 16, 256, 0, stream>>>(X16, W16b, bb, 3, feat16, nullptr, nullptr);
  gated_score_kernel<<<NN / 8, 256, 0, stream>>>(q16, feat16, cw, cb, Araw);
  pool_out_kernel<<<1, DD, 0, stream>>>(Araw, bufX1, outw, outb, (float*)d_out);
}
